// AFWM_Vitonhd_lrarms_87041807221027
// MI455X (gfx1250) — matrix operand order repaired and checked by reading, compiled, not run on hardware
//
#include <hip/hip_runtime.h>
#include <hip/hip_bf16.h>

// ---------------------------------------------------------------------------
// AFWM flow-estimation + warp for MI455X (gfx1250, wave32, WMMA, async->LDS).
// Fixed problem size from reference: B=8, C=256, H=128, W=96, stride=1.
//
// Pipeline:
//   1) corr_lrelu_kernel : 7x7 correlation (/C, lrelu) -> bf16 NHWC [B][H][W][64]
//   2) pack_weights      : OIHW fp32 -> bf16 [tap][cout_pad][cin_pad]
//   3) conv3x3_wmma      : per-(b,h) row block stages 3 rows (zero halo) into
//                          LDS via global_load_async_to_lds_b128, then all
//                          waves run implicit-GEMM v_wmma_f32_16x16x32_bf16
//                          from LDS A-fragments (ds_load_b128) + L2 B-fragments
//   4) warp_bilinear     : apply_offset + grid_sample fused, fp32 out (B,C,H,W)
// ---------------------------------------------------------------------------

typedef __attribute__((ext_vector_type(8)))  float  v8f;
typedef __attribute__((ext_vector_type(8)))  __bf16 v8bf;
typedef __attribute__((ext_vector_type(16))) __bf16 v16bf;

static __device__ __forceinline__ __bf16 f32_to_bf16(float f) {
  unsigned u = __builtin_bit_cast(unsigned, f);
  unsigned r = u + 0x7FFFu + ((u >> 16) & 1u);   // round-to-nearest-even
  unsigned short h = (unsigned short)(r >> 16);
  return __builtin_bit_cast(__bf16, h);
}

// Problem constants (reference setup_inputs)
#define BB 8
#define CC 256
#define HH 128
#define WW 96
#define NEG_SLOPE 0.1f

// ---------------------------------------------------------------------------
// Kernel 1: correlation + /C + lrelu, bf16 NHWC output with K padded to 64.
// Block = 96 threads (3 waves) handles one (b, h) row; loop over channels
// staging f1 row (96) and the 7x102 f2 window through LDS.
// ---------------------------------------------------------------------------
__global__ __launch_bounds__(96)
void corr_lrelu_kernel(const float* __restrict__ f1,
                       const float* __restrict__ f2,
                       __bf16* __restrict__ corr /* [B][H][W][64] */) {
  __shared__ float s1[WW];
  __shared__ float s2[7][WW + 8];        // need cols 0..101 (w-3 .. w+98)

  const int tid = threadIdx.x;           // 0..95, one output column each
  const int h   = blockIdx.x % HH;
  const int b   = blockIdx.x / HH;

  float acc[49];
#pragma unroll
  for (int d = 0; d < 49; ++d) acc[d] = 0.f;

  for (int c = 0; c < CC; ++c) {
    const float* f1p = f1 + ((size_t)(b * CC + c) * HH + h) * WW;
    s1[tid] = f1p[tid];
    const float* f2c = f2 + (size_t)(b * CC + c) * HH * WW;
    for (int i = tid; i < 7 * 102; i += 96) {
      int r  = i / 102;
      int cc = i % 102;
      int hr = h + r - 3;
      int wc = cc - 3;
      float v = 0.f;
      if (hr >= 0 && hr < HH && wc >= 0 && wc < WW) v = f2c[hr * WW + wc];
      s2[r][cc] = v;
    }
    __syncthreads();
    float a = s1[tid];
#pragma unroll
    for (int dy = 0; dy < 7; ++dy)
#pragma unroll
      for (int dx = 0; dx < 7; ++dx)
        acc[dy * 7 + dx] = fmaf(a, s2[dy][tid + dx], acc[dy * 7 + dx]);
    __syncthreads();
  }

  __bf16* o = corr + ((size_t)(b * HH + h) * WW + tid) * 64;
#pragma unroll
  for (int d = 0; d < 49; ++d) {
    float v = acc[d] * (1.f / (float)CC);
    v = (v > 0.f) ? v : NEG_SLOPE * v;
    o[d] = f32_to_bf16(v);
  }
#pragma unroll
  for (int d = 49; d < 64; ++d) o[d] = f32_to_bf16(0.f);
}

// ---------------------------------------------------------------------------
// Kernel 2: repack OIHW fp32 weights into bf16 [tap][cout_pad][cin_pad],
// zero-filled padding (so padded GEMM-K / GEMM-N contribute 0).
// ---------------------------------------------------------------------------
__global__ void pack_weights_kernel(const float* __restrict__ w,
                                    __bf16* __restrict__ out,
                                    int COUT, int CIN, int COUT_PAD, int CIN_PAD) {
  int idx = blockIdx.x * blockDim.x + threadIdx.x;
  int total = 9 * COUT_PAD * CIN_PAD;
  if (idx >= total) return;
  int c = idx % CIN_PAD;
  int n = (idx / CIN_PAD) % COUT_PAD;
  int t = idx / (CIN_PAD * COUT_PAD);
  float v = 0.f;
  if (n < COUT && c < CIN) v = w[(size_t)(n * CIN + c) * 9 + t];
  out[idx] = f32_to_bf16(v);
}

// ---------------------------------------------------------------------------
// Kernel 3: 3x3 same-pad conv as 9 shifted GEMMs.
//
// One 256-thread block (8 waves) per (b, h) output row:
//   Stage rows h-1..h+1, pixels -1..96 (zero halo) for all CIN into LDS with
//   per-lane 16B GLOBAL_LOAD_ASYNC_TO_LDS_B128 (ASYNCcnt), halo chunks
//   zero-filled with ds_store_b128. s_wait_asynccnt 0 + barrier, then each
//   wave walks (w-tile x n-tile) pairs: A-fragments from LDS (2x ds_load_b128
//   per lane), B-fragments from L2-resident packed weights (2x b128), one
//   v_wmma_f32_16x16x32_bf16 per 32-channel step. No divergence anywhere in
//   the compute phase -> EXEC stays all-ones around WMMA.
//
// Fragment layouts per CDNA5 ISA 7.12.2 (16-bit A 16x32 / B 32x16 / f32 C-D):
//   A: lane<16 -> M=lane, K in {0..7}u{16..23}; lane>=16 -> M=lane-16,
//      K in {8..15}u{24..31}  => two contiguous 8-half runs per lane.
//   B: N=lane%16, K as for A: {0..7}u{16..23} (lane<16) or {8..15}u{24..31}.
//   D: element r -> M = r + (lane<16?0:8), N = lane%16.
//
// LDS pixel stride padded by 8 halfs (16B) so the 16 lanes' ds_load_b128
// (pixel stride) walk different banks instead of all hitting one.
// ---------------------------------------------------------------------------
template <int CIN, int COUT_PAD, int COUT, bool LRELU, bool OUTF32>
__global__ __launch_bounds__(256)
void conv3x3_wmma_kernel(const __bf16* __restrict__ act,   // [B][H][W][CIN]
                         const __bf16* __restrict__ wpack, // [9][COUT_PAD][CIN]
                         const float* __restrict__ bias,   // [COUT]
                         __bf16* __restrict__ out_bf,      // [B][H][W][COUT_PAD]
                         float* __restrict__ out_f32) {    // [B][H][W][COUT]
  constexpr int PX     = WW + 2;          // 98 staged pixels (w = -1 .. 96)
  constexpr int CSTR   = CIN + 8;         // padded per-pixel stride (halfs)
  constexpr int NT     = COUT_PAD / 16;
  constexpr int NTILES = (WW / 16) * NT;  // (w-tile, n-tile) pairs per row
  constexpr int C8     = CIN / 8;         // 16B chunks per pixel

  extern __shared__ __bf16 smem[];        // [3][PX][CSTR]

  const int h   = blockIdx.x % HH;
  const int b   = blockIdx.x / HH;
  const int tid = threadIdx.x;

  // ---- stage 3 input rows (with zero halo) into LDS ----
  const int nchunk = 3 * PX * C8;
  for (int q = tid; q < nchunk; q += 256) {
    const int c8 = q % C8;
    const int p  = (q / C8) % PX;
    const int r  = q / (C8 * PX);
    const int hh = h + r - 1;
    const int wp = p - 1;
    __bf16* dst = smem + (size_t)(r * PX + p) * CSTR + c8 * 8;
    if (hh >= 0 && hh < HH && wp >= 0 && wp < WW) {
      const __bf16* src = act + ((size_t)(b * HH + hh) * WW + wp) * CIN + c8 * 8;
      // generic LDS address -> low 32 bits are the LDS byte offset
      unsigned ldsoff = (unsigned)(size_t)(void*)dst;
      asm volatile("global_load_async_to_lds_b128 %0, %1, off"
                   :: "v"(ldsoff), "v"(src)
                   : "memory");
    } else {
      v8bf z = {};
      *(v8bf*)dst = z;                    // zero halo via ds_store_b128
    }
  }
  asm volatile("s_wait_asynccnt 0x0" ::: "memory");
  __syncthreads();

  // ---- compute phase: 8 waves sweep the (w-tile, n-tile) grid ----
  const int  wave = tid >> 5;
  const int  lane = tid & 31;
  const int  lo   = lane & 15;
  const bool hi   = lane >= 16;

  for (int tl = wave; tl < NTILES; tl += 8) {
    const int wt = tl % (WW / 16);
    const int nt = (tl / (WW / 16)) * 16;
    const int w0 = wt * 16;

    v8f acc = {};
#pragma unroll
    for (int t = 0; t < 9; ++t) {
      const int dh = t / 3;               // staged row 0..2
      const int dw = t % 3;               // staged col offset 0..2
      // staged pixel index for A row M=lo: (w0 + lo + (dw-1)) + 1
      const __bf16* arow = smem + (size_t)(dh * PX + (w0 + lo + dw)) * CSTR;
      const __bf16* brow =
          wpack + ((size_t)(t * COUT_PAD + nt + lo)) * CIN + (hi ? 8 : 0);
#pragma unroll
      for (int c0 = 0; c0 < CIN; c0 += 32) {
        v8bf a0 = *(const v8bf*)(arow + c0 + (hi ? 8 : 0));
        v8bf a1 = *(const v8bf*)(arow + c0 + (hi ? 24 : 16));
        v8bf b0 = *(const v8bf*)(brow + c0);
        v8bf b1 = *(const v8bf*)(brow + c0 + 16);
        v16bf A, Bf;
#pragma unroll
        for (int i = 0; i < 8; ++i) {
          A[i] = a0[i];  A[8 + i] = a1[i];
          Bf[i] = b0[i]; Bf[8 + i] = b1[i];
        }
        acc = __builtin_amdgcn_wmma_f32_16x16x32_bf16(
            /*neg_a=*/false, A, /*neg_b=*/false, Bf,
            /*c_mod=*/(short)0, acc, /*reuse_a=*/false, /*reuse_b=*/false);
      }
    }

    const int   n  = nt + lo;
    const float bv = (n < COUT) ? bias[n] : 0.f;
#pragma unroll
    for (int r = 0; r < 8; ++r) {
      const int m    = r + (hi ? 8 : 0);
      const int wout = w0 + m;
      float v = acc[r] + bv;
      if (LRELU) v = (v > 0.f) ? v : NEG_SLOPE * v;
      const size_t pix = (size_t)(b * HH + h) * WW + wout;
      if (OUTF32) {
        if (n < COUT) out_f32[pix * COUT + n] = v;
      } else {
        out_bf[pix * COUT_PAD + n] = f32_to_bf16(v);
      }
    }
  }
}

// ---------------------------------------------------------------------------
// Kernel 4: fused apply_offset + bilinear grid_sample.
// One thread per (b,h,w); inner loop over C with coalesced loads along w.
// ---------------------------------------------------------------------------
__global__ __launch_bounds__(256)
void warp_bilinear_kernel(const float* __restrict__ img,   // feat2 NCHW fp32
                          const float* __restrict__ flow,  // [B][H][W][2]
                          float* __restrict__ out) {       // NCHW fp32
  int idx = blockIdx.x * blockDim.x + threadIdx.x;
  if (idx >= BB * HH * WW) return;
  const int w = idx % WW;
  const int h = (idx / WW) % HH;
  const int b = idx / (WW * HH);

  const float fx = flow[(size_t)idx * 2 + 0];
  const float fy = flow[(size_t)idx * 2 + 1];
  // grid math collapses to: sample at clamp(w+fx), clamp(h+fy)
  const float x = fminf(fmaxf((float)w + fx, 0.f), (float)(WW - 1));
  const float y = fminf(fmaxf((float)h + fy, 0.f), (float)(HH - 1));
  const float x0f = floorf(x), y0f = floorf(y);
  int x0 = (int)x0f, y0 = (int)y0f;
  int x1 = x0 + 1; if (x1 > WW - 1) x1 = WW - 1;
  int y1 = y0 + 1; if (y1 > HH - 1) y1 = HH - 1;
  const float wx = x - x0f, wy = y - y0f;
  const float w00 = (1.f - wx) * (1.f - wy), w01 = wx * (1.f - wy);
  const float w10 = (1.f - wx) * wy,         w11 = wx * wy;

  const int o00 = y0 * WW + x0, o01 = y0 * WW + x1;
  const int o10 = y1 * WW + x0, o11 = y1 * WW + x1;
  const int od  = h * WW + w;
  const float* base  = img + (size_t)b * CC * HH * WW;
  float*       obase = out + (size_t)b * CC * HH * WW;
  for (int c = 0; c < CC; ++c) {
    const float* p = base + (size_t)c * HH * WW;
    float v = p[o00] * w00 + p[o01] * w01 + p[o10] * w10 + p[o11] * w11;
    obase[(size_t)c * HH * WW + od] = v;
  }
}

// ---------------------------------------------------------------------------
// Host-side orchestration. Inputs (setup_inputs order):
//  0 feat1  1 feat2  2 w1  3 b1  4 w2  5 b2  6 w3  7 b3  8 w4  9 b4  10 stride(=1)
// ---------------------------------------------------------------------------
extern "C" void kernel_launch(void* const* d_in, const int* in_sizes, int n_in,
                              void* d_out, int out_size, void* d_ws, size_t ws_size,
                              hipStream_t stream) {
  const float* feat1 = (const float*)d_in[0];
  const float* feat2 = (const float*)d_in[1];
  const float* w1 = (const float*)d_in[2];  const float* b1 = (const float*)d_in[3];
  const float* w2 = (const float*)d_in[4];  const float* b2 = (const float*)d_in[5];
  const float* w3 = (const float*)d_in[6];  const float* b3 = (const float*)d_in[7];
  const float* w4 = (const float*)d_in[8];  const float* b4 = (const float*)d_in[9];
  float* out = (float*)d_out;
  (void)in_sizes; (void)n_in; (void)out_size; (void)ws_size; // stride==1 assumed

  const size_t PIX = (size_t)BB * HH * WW;     // 98304

  // Workspace carve-up (256-byte aligned chunks).
  char* ws = (char*)d_ws;
  size_t off = 0;
  auto carve = [&](size_t bytes) {
    void* p = ws + off;
    off += (bytes + 255) & ~(size_t)255;
    return p;
  };
  __bf16* corr = (__bf16*)carve(PIX * 64  * sizeof(__bf16));  // 12.6 MB
  __bf16* act1 = (__bf16*)carve(PIX * 128 * sizeof(__bf16));  // 25.2 MB
  __bf16* act2 = (__bf16*)carve(PIX * 64  * sizeof(__bf16));  // 12.6 MB
  __bf16* act3 = (__bf16*)carve(PIX * 32  * sizeof(__bf16));  //  6.3 MB
  float*  flow = (float*) carve(PIX * 2   * sizeof(float));   //  0.8 MB
  __bf16* wp1  = (__bf16*)carve((size_t)9 * 128 * 64  * sizeof(__bf16));
  __bf16* wp2  = (__bf16*)carve((size_t)9 * 64  * 128 * sizeof(__bf16));
  __bf16* wp3  = (__bf16*)carve((size_t)9 * 32  * 64  * sizeof(__bf16));
  __bf16* wp4  = (__bf16*)carve((size_t)9 * 16  * 32  * sizeof(__bf16));

  // Weight repacks (tiny, independent).
  pack_weights_kernel<<<(9 * 128 * 64  + 255) / 256, 256, 0, stream>>>(w1, wp1, 128, 49,  128, 64);
  pack_weights_kernel<<<(9 * 64  * 128 + 255) / 256, 256, 0, stream>>>(w2, wp2, 64,  128, 64,  128);
  pack_weights_kernel<<<(9 * 32  * 64  + 255) / 256, 256, 0, stream>>>(w3, wp3, 32,  64,  32,  64);
  pack_weights_kernel<<<(9 * 16  * 32  + 255) / 256, 256, 0, stream>>>(w4, wp4, 2,   32,  16,  32);

  // Correlation (+lrelu, /C) -> bf16 NHWC padded to 64 channels.
  corr_lrelu_kernel<<<BB * HH, 96, 0, stream>>>(feat1, feat2, corr);

  // Conv stack: one 8-wave block per (b,h) row; LDS = 3*(W+2)*(CIN+8)*2 bytes.
  const int ROWS = BB * HH;                                   // 1024 blocks
  conv3x3_wmma_kernel<64, 128, 128, true, false>
      <<<ROWS, 256, 3 * (WW + 2) * (64 + 8)  * 2, stream>>>(corr, wp1, b1, act1, nullptr);
  conv3x3_wmma_kernel<128, 64, 64, true, false>
      <<<ROWS, 256, 3 * (WW + 2) * (128 + 8) * 2, stream>>>(act1, wp2, b2, act2, nullptr);
  conv3x3_wmma_kernel<64, 32, 32, true, false>
      <<<ROWS, 256, 3 * (WW + 2) * (64 + 8)  * 2, stream>>>(act2, wp3, b3, act3, nullptr);
  conv3x3_wmma_kernel<32, 16, 2, false, true>
      <<<ROWS, 256, 3 * (WW + 2) * (32 + 8)  * 2, stream>>>(act3, wp4, b4, nullptr, flow);

  // Fused apply_offset + grid_sample.
  warp_bilinear_kernel<<<(int)((PIX + 255) / 256), 256, 0, stream>>>(feat2, flow, out);
}
